// PPIModel_37469294690488
// MI455X (gfx1250) — matrix operand order repaired and checked by reading, not compiled, not run on hardware
//
#include <hip/hip_runtime.h>
#include <hip/hip_bf16.h>

// ---------------------------------------------------------------------------
// PPI GraphConv model for MI455X (gfx1250), wave32 + WMMA bf16 path.
// ---------------------------------------------------------------------------

typedef __attribute__((ext_vector_type(16))) __bf16        v16bf;
typedef __attribute__((ext_vector_type(8)))  float         v8f;
typedef __attribute__((ext_vector_type(4)))  unsigned int  u32x4;

union FragBF {
    v16bf v;
    u32x4 q[2];
};

static constexpr int N_NODES = 100000;
static constexpr int N_EDGES = 1600000;
static constexpr int N_GRAPHS = 50;
static constexpr int IN_FEAT = 512;
static constexpr int HIDDEN  = 256;

__device__ __forceinline__ unsigned short f2bf(float f) {
    unsigned int u = __float_as_uint(f);
    unsigned int rounding = 0x7FFFu + ((u >> 16) & 1u);   // round-to-nearest-even
    u += rounding;
    return (unsigned short)(u >> 16);
}

// ---------------------------------------------------------------------------
// Degree accumulation: out_deg[src[e]] += 1 ; in_deg[dst[e]] += 1
// ---------------------------------------------------------------------------
__global__ void deg_kernel(const int* __restrict__ src, const int* __restrict__ dst,
                           float* __restrict__ out_deg, float* __restrict__ in_deg,
                           int n_edges) {
    int e = blockIdx.x * blockDim.x + threadIdx.x;
    if (e >= n_edges) return;
    atomicAdd(&out_deg[src[e]], 1.0f);
    atomicAdd(&in_deg[dst[e]], 1.0f);
}

// In-place: deg -> rsqrt(max(deg,1))
__global__ void norm_kernel(float* __restrict__ out_n, float* __restrict__ in_n, int n) {
    int t = blockIdx.x * blockDim.x + threadIdx.x;
    if (t >= n) return;
    out_n[t] = rsqrtf(fmaxf(out_n[t], 1.0f));
    in_n[t]  = rsqrtf(fmaxf(in_n[t], 1.0f));
}

// ---------------------------------------------------------------------------
// Convert W1 [512x256] f32 row-major -> W1T bf16 [256x512] (n-major, contiguous k)
// ---------------------------------------------------------------------------
__global__ void w1cvt_kernel(const float* __restrict__ W1, unsigned short* __restrict__ w1tb) {
    int t = blockIdx.x * blockDim.x + threadIdx.x;   // t = n*512 + k
    if (t >= IN_FEAT * HIDDEN) return;
    int n = t >> 9;
    int k = t & 511;
    w1tb[t] = f2bf(W1[k * HIDDEN + n]);
}

// ---------------------------------------------------------------------------
// GEMM1: h1[N,256] = (feat * out_norm) @ W1   via v_wmma_f32_16x16x32_bf16
// Block = 512 threads = 16 waves; block handles one 16-row M tile, each wave
// one 16-col N tile. A tile staged in LDS as bf16 with out_norm applied.
// ---------------------------------------------------------------------------
__global__ __launch_bounds__(512) void gemm1_kernel(
        const float* __restrict__ feat, const float* __restrict__ out_norm,
        const unsigned short* __restrict__ w1tb, float* __restrict__ h1) {
    __shared__ __align__(16) unsigned short As[16 * 520];   // 16 rows, stride 520 bf16

    const int t = threadIdx.x;
    const int mtile = blockIdx.x;                  // 0..6249

    // ---- stage A tile (16 x 512 f32 -> bf16, scaled by out_norm) ----
    {
        const int row = t >> 5;                    // 0..15
        const int c0  = (t & 31) * 16;             // 0..496, 16 elems per thread
        const int grow = mtile * 16 + row;
        const float nrm = out_norm[grow];
        const float4* s4 = (const float4*)(feat + (size_t)grow * IN_FEAT + c0);
        unsigned int* dp = (unsigned int*)(As + row * 520 + c0);
        #pragma unroll
        for (int i = 0; i < 4; ++i) {
            float4 v = s4[i];
            unsigned int lo = (unsigned int)f2bf(v.x * nrm) |
                              ((unsigned int)f2bf(v.y * nrm) << 16);
            unsigned int hi = (unsigned int)f2bf(v.z * nrm) |
                              ((unsigned int)f2bf(v.w * nrm) << 16);
            dp[i * 2 + 0] = lo;
            dp[i * 2 + 1] = hi;
        }
    }
    __syncthreads();

    // ---- per-wave WMMA over K = 512 in 16 steps of 32 ----
    const int wave = t >> 5;                       // n tile 0..15
    const int lane = t & 31;
    const int half = lane >> 4;                    // 0 | 1
    const int mn   = lane & 15;                    // row (A) / col (B) within tile
    const int ncol = wave * 16 + mn;

    v8f c = {0.f, 0.f, 0.f, 0.f, 0.f, 0.f, 0.f, 0.f};

    const unsigned short* brow = w1tb + (size_t)ncol * IN_FEAT;

    #pragma unroll
    for (int kk = 0; kk < 16; ++kk) {
        const int k0 = kk * 32;
        FragBF a, b;
        // A 16x32 bf16 layout: lane(mn,half): e<8 -> K=k0+8*half+e ; e>=8 -> K=k0+16+8*half+(e-8)
        a.q[0] = *(const u32x4*)(As + mn * 520 + k0 + 8 * half);
        a.q[1] = *(const u32x4*)(As + mn * 520 + k0 + 16 + 8 * half);
        // B 32x16 bf16 layout: lane(n,half): e<8 -> K=k0+8*half+e ; e>=8 -> K=k0+16+8*half+(e-8)
        b.q[0] = *(const u32x4*)(brow + k0 + 8 * half);
        b.q[1] = *(const u32x4*)(brow + k0 + 16 + 8 * half);
        c = __builtin_amdgcn_wmma_f32_16x16x32_bf16(
                /*neg_a=*/false, a.v, /*neg_b=*/false, b.v,
                /*c_mod=*/(short)0, c, /*reuse_a=*/false, /*reuse_b=*/false);
    }

    // ---- store C: VGPR r -> (M = r + 8*half, N = mn) ----
    float* out = h1 + (size_t)(mtile * 16) * HIDDEN + wave * 16 + mn;
    #pragma unroll
    for (int r = 0; r < 8; ++r) {
        out[(size_t)(r + 8 * half) * HIDDEN] = c[r];
    }
}

// ---------------------------------------------------------------------------
// Scatter 1: agg1[dst[e], :] += h1[src[e], :]   (256 floats/edge, 64 thr/edge)
// ---------------------------------------------------------------------------
__global__ void scatter1_kernel(const float* __restrict__ h1,
                                const int* __restrict__ src, const int* __restrict__ dst,
                                float* __restrict__ agg1, int n_edges) {
    long long t = (long long)blockIdx.x * blockDim.x + threadIdx.x;
    int e = (int)(t >> 6);
    if (e >= n_edges) return;
    int f = ((int)t & 63) * 4;
    int s = src[e], d = dst[e];
    float4 v = *(const float4*)(h1 + (size_t)s * HIDDEN + f);
    float* o = agg1 + (size_t)d * HIDDEN + f;
    atomicAdd(o + 0, v.x);
    atomicAdd(o + 1, v.y);
    atomicAdd(o + 2, v.z);
    atomicAdd(o + 3, v.w);
}

// x1 = relu(agg1 * in_norm + b1), in place; float4 per thread
__global__ void post1_kernel(float* __restrict__ agg1, const float* __restrict__ in_norm,
                             const float* __restrict__ b1, int n4) {
    int t = blockIdx.x * blockDim.x + threadIdx.x;
    if (t >= n4) return;
    int idx = t * 4;
    int row = idx >> 8;
    int col = idx & 255;
    float nn = in_norm[row];
    float4* p = (float4*)agg1 + t;
    float4 v = *p;
    v.x = fmaxf(v.x * nn + b1[col + 0], 0.f);
    v.y = fmaxf(v.y * nn + b1[col + 1], 0.f);
    v.z = fmaxf(v.z * nn + b1[col + 2], 0.f);
    v.w = fmaxf(v.w * nn + b1[col + 3], 0.f);
    *p = v;
}

// ---------------------------------------------------------------------------
// GEMM2: h2[N,2] = (x1 * out_norm) @ W2 [256x2]  (thread per node; W2 in LDS)
// ---------------------------------------------------------------------------
__global__ __launch_bounds__(256) void gemm2_kernel(
        const float* __restrict__ x1, const float* __restrict__ out_norm,
        const float* __restrict__ W2, float* __restrict__ h2, int n) {
    __shared__ float w2s[512];
    int t = threadIdx.x;
    w2s[t] = W2[t];
    w2s[t + 256] = W2[t + 256];
    __syncthreads();
    int node = blockIdx.x * blockDim.x + t;
    if (node >= n) return;
    const float4* xr = (const float4*)(x1 + (size_t)node * HIDDEN);
    float a0 = 0.f, a1 = 0.f;
    #pragma unroll 8
    for (int k4 = 0; k4 < 64; ++k4) {
        float4 v = xr[k4];
        int k2 = k4 * 8;   // 2*k
        a0 += v.x * w2s[k2 + 0] + v.y * w2s[k2 + 2] + v.z * w2s[k2 + 4] + v.w * w2s[k2 + 6];
        a1 += v.x * w2s[k2 + 1] + v.y * w2s[k2 + 3] + v.z * w2s[k2 + 5] + v.w * w2s[k2 + 7];
    }
    float nn = out_norm[node];
    h2[(size_t)node * 2 + 0] = a0 * nn;
    h2[(size_t)node * 2 + 1] = a1 * nn;
}

// Scatter 2: agg2[dst[e], :2] += h2[src[e], :2]
__global__ void scatter2_kernel(const float* __restrict__ h2,
                                const int* __restrict__ src, const int* __restrict__ dst,
                                float* __restrict__ agg2, int n_edges) {
    int e = blockIdx.x * blockDim.x + threadIdx.x;
    if (e >= n_edges) return;
    int s = src[e], d = dst[e];
    float2 v = *(const float2*)(h2 + (size_t)s * 2);
    atomicAdd(&agg2[(size_t)d * 2 + 0], v.x);
    atomicAdd(&agg2[(size_t)d * 2 + 1], v.y);
}

// x2 = relu(agg2 * in_norm + b2) fused with per-graph avg/max readout
__global__ void readout_kernel(const float* __restrict__ agg2, const float* __restrict__ in_norm,
                               const float* __restrict__ b2, const int* __restrict__ gid,
                               float* __restrict__ gsum, float* __restrict__ gmax,
                               float* __restrict__ gcnt, int n) {
    int node = blockIdx.x * blockDim.x + threadIdx.x;
    if (node >= n) return;
    float nn = in_norm[node];
    float x0 = fmaxf(agg2[(size_t)node * 2 + 0] * nn + b2[0], 0.f);
    float x1 = fmaxf(agg2[(size_t)node * 2 + 1] * nn + b2[1], 0.f);
    int g = gid[node];
    atomicAdd(&gsum[g * 2 + 0], x0);
    atomicAdd(&gsum[g * 2 + 1], x1);
    atomicAdd(&gcnt[g], 1.0f);
    // ReLU outputs >= 0 -> uint ordering == float ordering (max init = 0)
    atomicMax((unsigned int*)&gmax[g * 2 + 0], __float_as_uint(x0));
    atomicMax((unsigned int*)&gmax[g * 2 + 1], __float_as_uint(x1));
}

// pooled = [avg0, avg1, max0, max1]; out = sigmoid(pooled . fcW + fcb)
__global__ void final_kernel(const float* __restrict__ gsum, const float* __restrict__ gmax,
                             const float* __restrict__ gcnt, const float* __restrict__ fcW,
                             const float* __restrict__ fcb, float* __restrict__ out, int G) {
    int g = blockIdx.x * blockDim.x + threadIdx.x;
    if (g >= G) return;
    float cnt = fmaxf(gcnt[g], 1.0f);
    float z = (gsum[g * 2 + 0] / cnt) * fcW[0] +
              (gsum[g * 2 + 1] / cnt) * fcW[1] +
              gmax[g * 2 + 0] * fcW[2] +
              gmax[g * 2 + 1] * fcW[3] + fcb[0];
    out[g] = 1.0f / (1.0f + expf(-z));
}

// ---------------------------------------------------------------------------
extern "C" void kernel_launch(void* const* d_in, const int* in_sizes, int n_in,
                              void* d_out, int out_size, void* d_ws, size_t ws_size,
                              hipStream_t stream) {
    const float* feat = (const float*)d_in[0];
    const int*   src  = (const int*)d_in[1];
    const int*   dst  = (const int*)d_in[2];
    const int*   gid  = (const int*)d_in[3];
    const float* W1   = (const float*)d_in[4];
    const float* b1   = (const float*)d_in[5];
    const float* W2   = (const float*)d_in[6];
    const float* b2   = (const float*)d_in[7];
    const float* fcW  = (const float*)d_in[8];
    const float* fcb  = (const float*)d_in[9];
    float* out = (float*)d_out;

    const size_t N = N_NODES, E = N_EDGES;

    // workspace layout (floats)
    float* ws       = (float*)d_ws;
    float* out_norm = ws;                     // N  (degree -> norm in place)
    float* in_norm  = ws + N;                 // N
    float* h1       = ws + 2 * N;             // N*256
    float* agg1     = h1 + N * 256;           // N*256 (becomes x1)
    float* h2       = agg1 + N * 256;         // N*2
    float* agg2     = h2 + 2 * N;             // N*2
    float* gsum     = agg2 + 2 * N;           // 128 (50*2 used)
    float* gmax     = gsum + 128;             // 128
    float* gcnt     = gmax + 128;             // 128
    unsigned short* w1tb = (unsigned short*)(gcnt + 128);   // 256*512 bf16

    // zero-init accumulators (graph-capturable memsets)
    hipMemsetAsync(out_norm, 0, 2 * N * sizeof(float), stream);
    hipMemsetAsync(agg1, 0, N * 256 * sizeof(float), stream);
    hipMemsetAsync(agg2, 0, 2 * N * sizeof(float), stream);
    hipMemsetAsync(gsum, 0, 3 * 128 * sizeof(float), stream);

    // degrees -> norms
    deg_kernel<<<(E + 255) / 256, 256, 0, stream>>>(src, dst, out_norm, in_norm, (int)E);
    norm_kernel<<<(N + 255) / 256, 256, 0, stream>>>(out_norm, in_norm, (int)N);

    // weight conversion + layer 1 GEMM (WMMA bf16)
    w1cvt_kernel<<<(IN_FEAT * HIDDEN + 255) / 256, 256, 0, stream>>>(W1, w1tb);
    gemm1_kernel<<<N_NODES / 16, 512, 0, stream>>>(feat, out_norm, w1tb, h1);

    // layer 1 scatter + bias/relu
    scatter1_kernel<<<(unsigned)((E * 64) / 256), 256, 0, stream>>>(h1, src, dst, agg1, (int)E);
    post1_kernel<<<(unsigned)((N * 256 / 4 + 255) / 256), 256, 0, stream>>>(agg1, in_norm, b1,
                                                                            (int)(N * 256 / 4));

    // layer 2 GEMM + scatter + fused relu/readout
    gemm2_kernel<<<(N + 255) / 256, 256, 0, stream>>>(agg1, out_norm, W2, h2, (int)N);
    scatter2_kernel<<<(E + 255) / 256, 256, 0, stream>>>(h2, src, dst, agg2, (int)E);
    readout_kernel<<<(N + 255) / 256, 256, 0, stream>>>(agg2, in_norm, b2, gid,
                                                        gsum, gmax, gcnt, (int)N);

    // final FC + sigmoid
    final_kernel<<<1, 64, 0, stream>>>(gsum, gmax, gcnt, fcW, fcb, out, N_GRAPHS);
}
